// BinConv3d_27986006901372
// MI455X (gfx1250) — matrix operand order repaired and checked by reading, not compiled, not run on hardware
//
#include <hip/hip_runtime.h>
#include <stdint.h>

// ---------------------------------------------------------------------------
// Binarized 3D conv, NCDHW:  out = conv3d(sign(x), W, pad=1) + b
//   x: (1,32,64,128,128) f32   W: (64,32,3,3,3) f32   b: (64,) f32
//   out: (1,64,64,128,128) f32
//
// Implicit GEMM on v_wmma_f32_16x16x32_f16:
//   M = spatial (16 consecutive w), N = 64 output channels, K = 27 taps x 32 ic
// One 256-thread block (8 waves) handles (d, 4 h-rows); sign(x) halo tile and
// B-layout f16 weights are staged in LDS (weights via TDM when available).
// ---------------------------------------------------------------------------

typedef __attribute__((ext_vector_type(16))) _Float16     v16h;
typedef __attribute__((ext_vector_type(8)))  float        v8f;
typedef __attribute__((ext_vector_type(4)))  unsigned int v4u;
typedef __attribute__((ext_vector_type(8)))  int          v8i;
typedef __attribute__((ext_vector_type(4)))  int          v4i;

#define C_IN   32
#define C_OUT  64
#define DD     64
#define HH     128
#define WW     128
#define H_TILE 4
#define XSTRIDE 132                                   // padded wx extent (wx in [0,129])
#define X_HALVES (3*(H_TILE+2)*XSTRIDE*C_IN)          // 76032 halves
#define X_BYTES  (X_HALVES*2)                         // 152064 B
#define WB_HALVES (27*4*32*16)                        // 55296 halves (tap,ntile,lane,j)
#define WB_BYTES  (WB_HALVES*2)                       // 110592 B
#define OUT_FLOATS (C_OUT*WW)                         // 8192
#define LDS_TOTAL (X_BYTES + WB_BYTES + OUT_FLOATS*4) // 295424 B  (<= 320 KB/WGP)

union AV { v16h v; v4u q[2]; };

// ---- prep: W (OIDHW f32) -> f16 in exact WMMA-B register layout ------------
// ws[((tap*4+nt)*32 + lane)*16 + j] = f16(W[oc=nt*16+(lane&15)][ic=(j&8)*2+(lane>>4)*8+(j&7)][tap])
__global__ __launch_bounds__(256) void binconv_prep_w(const float* __restrict__ W,
                                                      _Float16* __restrict__ wb) {
    int i = blockIdx.x * 256 + threadIdx.x;
    if (i >= WB_HALVES) return;
    int j   = i & 15;
    int L   = (i >> 4) & 31;
    int nt  = (i >> 9) & 3;
    int tap = i >> 11;
    int oc = nt * 16 + (L & 15);
    int ic = (j & 8) * 2 + (L >> 4) * 8 + (j & 7);
    int kd = tap / 9, kh = (tap / 3) % 3, kw = tap % 3;
    float v = W[(((oc * C_IN + ic) * 3 + kd) * 3 + kh) * 3 + kw];
    wb[i] = (_Float16)v;
}

// ---- main kernel -----------------------------------------------------------
__global__ __launch_bounds__(256) void binconv_main(const float* __restrict__ x,
                                                    const _Float16* __restrict__ wb,
                                                    const float* __restrict__ bias,
                                                    float* __restrict__ out) {
    extern __shared__ char smem[];                 // [x signs | B weights | out tile]
    const int tid  = threadIdx.x;
    const int lane = tid & 31;
    const int wave = tid >> 5;
    const int d    = blockIdx.x >> 5;              // 64 d slices
    const int hb   = (blockIdx.x & 31) * H_TILE;   // 32 h-blocks of 4 rows

    // ---- stage weights (global d_ws -> LDS) --------------------------------
#if defined(__HIP_DEVICE_COMPILE__) && __has_builtin(__builtin_amdgcn_tensor_load_to_lds) && __has_builtin(__builtin_amdgcn_s_wait_tensorcnt)
    if (wave == 0) {
        // Tensor DMA descriptor: 1-D tile of WB_BYTES/8 8-byte elements.
        const unsigned long long n8 = WB_BYTES / 8;          // 13824
        unsigned long long g0q[2], g1q[4];
        unsigned long long gaddr = (unsigned long long)(uintptr_t)wb;
        g0q[0] = 1ull                                        // count = 1 valid descriptor
               | ((unsigned long long)(unsigned)X_BYTES << 32); // lds_addr
        g0q[1] = (gaddr & 0x01FFFFFFFFFFFFFFull)             // global_addr [56:0]
               | (2ull << 62);                               // type = 2 ("image")
        g1q[0] = (3ull << 16)                                // data_size = 3 (8 B)
               | (n8 << 48);                                 // tensor_dim0 lo16
        g1q[1] = (n8 >> 16)                                  // tensor_dim0 hi16
               | (1ull << 16)                                // tensor_dim1 = 1
               | (n8 << 48);                                 // tile_dim0
        g1q[2] = 1ull                                        // tile_dim1 = 1
               | (n8 << 32);                                 // tensor_dim0_stride lo32
        g1q[3] = 0ull;                                       // stride hi / dim1 stride
        v4u g0; v8i g1; v4i g2 = {0, 0, 0, 0}, g3 = {0, 0, 0, 0};
        __builtin_memcpy(&g0, g0q, 16);
        __builtin_memcpy(&g1, g1q, 32);
  #if __has_include(<hip/amd_detail/amd_gfx1250_TDM.h>)
        v8i g4 = {0, 0, 0, 0, 0, 0, 0, 0};                   // clang-23 6-arg form
        __builtin_amdgcn_tensor_load_to_lds(g0, g1, g2, g3, g4, 0);
  #else
        __builtin_amdgcn_tensor_load_to_lds(g0, g1, g2, g3, 0); // ROCm 7.2 5-arg form
  #endif
        __builtin_amdgcn_s_wait_tensorcnt(0);
    }
#else
    {
        v4u* dst = (v4u*)(smem + X_BYTES);
        const v4u* src = (const v4u*)wb;
        for (int i = tid; i < WB_BYTES / 16; i += 256) dst[i] = src[i];
    }
#endif

    // ---- stage sign(x) halo tile into LDS as +/-1 f16 ----------------------
    // LDS layout: xs[((dz*(H_TILE+2)+hy)*XSTRIDE + wx)*32 + ic], wx = wglob+1
    {
        _Float16* xs = (_Float16*)smem;
        const int NX = 3 * (H_TILE + 2) * 130 * C_IN;        // 74880
        for (int i = tid; i < NX; i += 256) {
            int wx = i % 130;
            int t  = i / 130;
            int ic = t % C_IN;  t /= C_IN;
            int hy = t % (H_TILE + 2);
            int dz = t / (H_TILE + 2);
            int dg = d + dz - 1, hg = hb + hy - 1, wg = wx - 1;
            float v = 0.0f;
            if ((unsigned)dg < DD && (unsigned)hg < HH && (unsigned)wg < WW) {
                size_t gi = (((size_t)ic * DD + dg) * HH + hg) * WW + wg;
                v = x[gi];
                __builtin_prefetch(x + gi + 2048, 0, 1);     // global_prefetch_b8
            }
            _Float16 s = (v > 0.0f) ? (_Float16)1.0f
                                    : ((v < 0.0f) ? (_Float16)-1.0f : (_Float16)0.0f);
            xs[((dz * (H_TILE + 2) + hy) * XSTRIDE + wx) * C_IN + ic] = s;
        }
    }
    __syncthreads();

    // ---- implicit GEMM: 27 taps x 16 WMMAs (4 h-rows x 4 oc-tiles) ---------
    const int wbase = wave * 16;          // this wave's 16-w M tile
    const int m     = lane & 15;          // M row within tile
    const int hi    = lane >> 4;          // lane half-group (K split)

    v8f acc[H_TILE][4];
#pragma unroll
    for (int hl = 0; hl < H_TILE; ++hl)
#pragma unroll
        for (int nt = 0; nt < 4; ++nt)
            acc[hl][nt] = (v8f){0.f, 0.f, 0.f, 0.f, 0.f, 0.f, 0.f, 0.f};

    for (int tap = 0; tap < 27; ++tap) {
        const int kd = tap / 9, kh = (tap / 3) % 3, kw = tap % 3;

        AV A[H_TILE];
#pragma unroll
        for (int hl = 0; hl < H_TILE; ++hl) {
            // A halves j: K = (j<8 ? 0 : 16) + 8*hi + (j&7)  -> two 16B chunks 32B apart
            const char* p = smem
                + (size_t)(((kd * (H_TILE + 2) + (hl + kh)) * XSTRIDE + (wbase + m + kw)) * C_IN) * 2
                + hi * 16;
            A[hl].q[0] = *(const v4u*)p;
            A[hl].q[1] = *(const v4u*)(p + 32);
        }
#pragma unroll
        for (int nt = 0; nt < 4; ++nt) {
            AV B;
            const char* pb = smem + X_BYTES + (size_t)(((tap * 4 + nt) * 32 + lane) * 32);
            B.q[0] = *(const v4u*)pb;
            B.q[1] = *(const v4u*)(pb + 16);
#pragma unroll
            for (int hl = 0; hl < H_TILE; ++hl) {
                acc[hl][nt] = __builtin_amdgcn_wmma_f32_16x16x32_f16(
                    false, A[hl].v, false, B.v, (short)0, acc[hl][nt], false, false);
            }
        }
    }

    // ---- epilogue: transpose via LDS for coalesced stores, add bias --------
    float* os = (float*)(smem + X_BYTES + WB_BYTES);
    for (int hl = 0; hl < H_TILE; ++hl) {
        __syncthreads();                       // prior store loop done with os
#pragma unroll
        for (int nt = 0; nt < 4; ++nt) {
            int oc = nt * 16 + m;              // C/D layout: lane -> N, vgpr r -> M
#pragma unroll
            for (int r = 0; r < 8; ++r) {
                int w = wbase + r + 8 * hi;
                os[oc * WW + w] = acc[hl][nt][r];
            }
        }
        __syncthreads();
        const int hg = hb + hl;
        for (int i = tid; i < OUT_FLOATS; i += 256) {
            int oc = i >> 7, w = i & 127;
            out[(((size_t)oc * DD + d) * HH + hg) * WW + w] = os[i] + bias[oc];
        }
    }
}

// ---------------------------------------------------------------------------
extern "C" void kernel_launch(void* const* d_in, const int* in_sizes, int n_in,
                              void* d_out, int out_size, void* d_ws, size_t ws_size,
                              hipStream_t stream) {
    (void)in_sizes; (void)n_in; (void)out_size; (void)ws_size;
    const float* x = (const float*)d_in[0];
    const float* W = (const float*)d_in[1];
    const float* b = (const float*)d_in[2];
    float*     out = (float*)d_out;
    _Float16*   wb = (_Float16*)d_ws;           // f16 weights in WMMA-B layout

    binconv_prep_w<<<(WB_HALVES + 255) / 256, 256, 0, stream>>>(W, wb);
    binconv_main<<<DD * (HH / H_TILE), 256, LDS_TOTAL, stream>>>(x, wb, b, out);
}
